// MultiHeadAttention_19250043421425
// MI455X (gfx1250) — matrix operand order repaired and checked by reading, not compiled, not run on hardware
//
#include <hip/hip_runtime.h>
#include <hip/hip_bf16.h>

// ---------------------------------------------------------------------------
// MHA forward on gfx1250 (wave32, WMMA bf16 -> f32 accum, async LDS staging)
//   Qp = q@Wq+bq  (bf16 [B*S,128])
//   Kp = k@Wk+bk  (bf16 [B*S,128])
//   Vt = (v@Wv+bv) stored head-transposed bf16 [B*H][32][S]
//   flash attention per (b,h): 64-key chunks double-buffered into LDS via
//   global_load_async_to_lds_b128, 16-query tiles per wave
//   out = Cc@Wo+bo (f32)
// ---------------------------------------------------------------------------

typedef __bf16 bf16;
typedef __attribute__((ext_vector_type(16))) __bf16 v16bf;
typedef __attribute__((ext_vector_type(8)))  __bf16 v8bf;
typedef __attribute__((ext_vector_type(8)))  float  v8f;

union V16 { v16bf v; v8bf h[2]; };

static __device__ inline v8f wmma_bf16(v16bf a, v16bf b, v8f c) {
  // D = A(16x32 bf16) * B(32x16 bf16) + C(16x16 f32)
  return __builtin_amdgcn_wmma_f32_16x16x32_bf16(
      /*neg_a=*/false, a, /*neg_b=*/false, b,
      /*c_mod=*/(short)0, c, /*reuse_a=*/false, /*reuse_b=*/false);
}

// LDS byte offset from a flat pointer to __shared__ (aperture rule: addr[31:0])
static __device__ inline unsigned lds_off(const void* p) {
  return (unsigned)(unsigned long long)p;
}

// 16B async copy global -> LDS (ASYNCcnt-tracked, per-lane)
static __device__ inline void async_b128(const void* gptr, void* lptr) {
  asm volatile("global_load_async_to_lds_b128 %0, %1, off"
               :: "v"(lds_off(lptr)), "v"((unsigned long long)gptr)
               : "memory");
}

#define DM 128
#define NH 4

// out_mode: 0 = bf16 row-major [M,128]
//           1 = f32  row-major [M,128]
//           2 = bf16 head-transposed Vt[(b*NH+h)*32+d][s]
__global__ __launch_bounds__(256) void gemm128_kernel(
    const float* __restrict__ Xf, const bf16* __restrict__ Xb,
    const float* __restrict__ W, const float* __restrict__ bias,
    bf16* __restrict__ outb, float* __restrict__ outf,
    int out_mode, int S)
{
  __shared__ __align__(16) bf16 wt[128][136]; // W transposed, padded

  for (int i = threadIdx.x; i < 128 * 128; i += blockDim.x) {
    int k = i >> 7, n = i & 127;
    wt[n][k] = (bf16)W[i];          // coalesced f32 read, transpose into LDS
  }
  __syncthreads();

  const int wave = threadIdx.x >> 5;
  const int lane = threadIdx.x & 31;
  const int half = lane >> 4;
  const int l16  = lane & 15;
  const int m0   = blockIdx.x * 128 + wave * 16;
  const int mrow = m0 + l16;

  // A fragments: 4 K-chunks of 32 (16-bit A layout, ISA 7.12.2)
  V16 a[4];
  if (Xf) {
    const float* xr = Xf + (size_t)mrow * DM;
#pragma unroll
    for (int c = 0; c < 4; ++c) {
      int base = c * 32 + half * 8;
#pragma unroll
      for (int e = 0; e < 8; ++e) {
        a[c].v[e]     = (bf16)xr[base + e];
        a[c].v[8 + e] = (bf16)xr[base + 16 + e];
      }
    }
  } else {
    const bf16* xr = Xb + (size_t)mrow * DM;
#pragma unroll
    for (int c = 0; c < 4; ++c) {
      int base = c * 32 + half * 8;
      a[c].h[0] = *(const v8bf*)(xr + base);
      a[c].h[1] = *(const v8bf*)(xr + base + 16);
    }
  }

#pragma unroll
  for (int nt = 0; nt < 8; ++nt) {
    const int n0 = nt * 16;
    const int n  = n0 + l16;
    const float bv = bias[n];
    v8f acc;
#pragma unroll
    for (int v = 0; v < 8; ++v) acc[v] = bv;

#pragma unroll
    for (int c = 0; c < 4; ++c) {
      V16 bfrag;
      const bf16* wr = &wt[n][c * 32 + half * 8];
      bfrag.h[0] = *(const v8bf*)wr;
      bfrag.h[1] = *(const v8bf*)(wr + 16);
      acc = wmma_bf16(a[c].v, bfrag.v, acc);
    }

    if (out_mode == 0) {
#pragma unroll
      for (int v = 0; v < 8; ++v)
        outb[(size_t)(m0 + half * 8 + v) * DM + n] = (bf16)acc[v];
    } else if (out_mode == 1) {
#pragma unroll
      for (int v = 0; v < 8; ++v)
        outf[(size_t)(m0 + half * 8 + v) * DM + n] = acc[v];
    } else {
      const int hh = n >> 5, dd = n & 31;
      const int r0 = m0 + half * 8;
      const int bb = r0 / S, ss = r0 % S;
      v8bf pk;
#pragma unroll
      for (int v = 0; v < 8; ++v) pk[v] = (bf16)acc[v];
      *(v8bf*)(outb + ((size_t)(bb * NH + hh) * 32 + dd) * S + ss) = pk;
    }
  }
}

// LDS row paddings (bf16 elements); strides 80B / 144B: 16B-aligned and
// bank-conflict-free across the 16-lane fragment reads.
#define KPAD 40
#define VPAD 72
#define PPAD 72

__global__ __launch_bounds__(256) void flash_attn_kernel(
    const bf16* __restrict__ Qp, const bf16* __restrict__ Kp,
    const bf16* __restrict__ Vt, bf16* __restrict__ Cc, int S)
{
  __shared__ __align__(16) bf16 kbuf[2][64][KPAD]; // K chunk: 64 keys x 32 d
  __shared__ __align__(16) bf16 vbuf[2][32][VPAD]; // V chunk: 32 d x 64 keys
  __shared__ __align__(16) bf16 pst[8][16][PPAD];  // per-wave P: 16 q x 64 keys

  const int tid  = threadIdx.x;
  const int wave = tid >> 5;
  const int lane = tid & 31;
  const int half = lane >> 4;
  const int l16  = lane & 15;
  const int bh   = blockIdx.x;          // b*NH + h
  const int b    = bh >> 2, h = bh & 3;
  const int q0   = blockIdx.y * 128 + wave * 16;
  const float scale = 0.17677669529663687f;   // 1/sqrt(32)

  // cooperative staging: each thread copies one 16B K segment + one 16B V segment
  const bf16* Kbase = Kp + (size_t)b * S * DM + h * 32;
  const bf16* Vbase = Vt + (size_t)bh * 32 * S;
  const int kkey = tid >> 2, kseg = tid & 3;   // 64 keys x 4 segs (8 bf16)
  const int vd   = tid >> 3, vseg = tid & 7;   // 32 d    x 8 segs (8 bf16)

  // Q A-fragment: 16 rows x depth 32 (whole head depth)
  V16 qf;
  {
    const bf16* qr = Qp + ((size_t)(b * S + q0 + l16) * DM) + h * 32;
    const int eo = half * 8;
    qf.h[0] = *(const v8bf*)(qr + eo);
    qf.h[1] = *(const v8bf*)(qr + 16 + eo);
  }

  float mrow[8], lrow[8];
  v8f acc0 = {}, acc1 = {};
#pragma unroll
  for (int v = 0; v < 8; ++v) { mrow[v] = -1e30f; lrow[v] = 0.0f; }

  const int nch = S / 64;

  // prologue: stage chunk 0
  async_b128(Kbase + (size_t)kkey * DM + kseg * 8, &kbuf[0][kkey][kseg * 8]);
  async_b128(Vbase + (size_t)vd * S + vseg * 8,    &vbuf[0][vd][vseg * 8]);

  for (int c = 0; c < nch; ++c) {
    const int cur = c & 1;
    if (c + 1 < nch) {
      const int nxt = cur ^ 1;
      const int k0n = (c + 1) * 64;
      async_b128(Kbase + (size_t)(k0n + kkey) * DM + kseg * 8, &kbuf[nxt][kkey][kseg * 8]);
      async_b128(Vbase + (size_t)vd * S + k0n + vseg * 8,      &vbuf[nxt][vd][vseg * 8]);
      asm volatile("s_wait_asynccnt 0x2" ::: "memory"); // current chunk landed
    } else {
      asm volatile("s_wait_asynccnt 0x0" ::: "memory");
    }
    __syncthreads();

    // logits: 4 key sub-tiles of 16 (Kdim = depth 32)
    v8f s[4];
#pragma unroll
    for (int j = 0; j < 4; ++j) {
      V16 kf;
      const bf16* kr = &kbuf[cur][j * 16 + l16][half * 8];
      kf.h[0] = *(const v8bf*)kr;
      kf.h[1] = *(const v8bf*)(kr + 16);
      v8f z = {};
      s[j] = wmma_bf16(qf.v, kf.v, z);
    }

    // online softmax over 64 keys; row = v + 8*half
    float f[8];
#pragma unroll
    for (int v = 0; v < 8; ++v) {
      const float a0 = s[0][v] * scale, a1 = s[1][v] * scale;
      const float a2 = s[2][v] * scale, a3 = s[3][v] * scale;
      float cm = fmaxf(fmaxf(a0, a1), fmaxf(a2, a3));
      cm = fmaxf(cm, __shfl_xor(cm, 1, 32));
      cm = fmaxf(cm, __shfl_xor(cm, 2, 32));
      cm = fmaxf(cm, __shfl_xor(cm, 4, 32));
      cm = fmaxf(cm, __shfl_xor(cm, 8, 32));
      const float mn = fmaxf(mrow[v], cm);
      const float p0 = __expf(a0 - mn), p1 = __expf(a1 - mn);
      const float p2 = __expf(a2 - mn), p3 = __expf(a3 - mn);
      float rs = (p0 + p1) + (p2 + p3);
      rs += __shfl_xor(rs, 1, 32);
      rs += __shfl_xor(rs, 2, 32);
      rs += __shfl_xor(rs, 4, 32);
      rs += __shfl_xor(rs, 8, 32);
      const float fv = __expf(mrow[v] - mn);
      lrow[v] = lrow[v] * fv + rs;
      mrow[v] = mn;
      f[v] = fv;
      bf16* pw = &pst[wave][half * 8 + v][l16];
      pw[0]  = (bf16)p0;
      pw[16] = (bf16)p1;
      pw[32] = (bf16)p2;
      pw[48] = (bf16)p3;
    }
#pragma unroll
    for (int v = 0; v < 8; ++v) { acc0[v] *= f[v]; acc1[v] *= f[v]; }

    // P(16x64) @ V(64x32): two 32-key A fragments x two 16-depth B tiles
#pragma unroll
    for (int fgr = 0; fgr < 2; ++fgr) {
      V16 pf;
      const bf16* pr = &pst[wave][l16][fgr * 32 + half * 8];
      pf.h[0] = *(const v8bf*)pr;
      pf.h[1] = *(const v8bf*)(pr + 16);
      V16 vf0, vf1;
      const bf16* vr0 = &vbuf[cur][l16][fgr * 32 + half * 8];
      vf0.h[0] = *(const v8bf*)vr0;
      vf0.h[1] = *(const v8bf*)(vr0 + 16);
      const bf16* vr1 = &vbuf[cur][16 + l16][fgr * 32 + half * 8];
      vf1.h[0] = *(const v8bf*)vr1;
      vf1.h[1] = *(const v8bf*)(vr1 + 16);
      acc0 = wmma_bf16(pf.v, vf0.v, acc0);
      acc1 = wmma_bf16(pf.v, vf1.v, acc1);
    }
    __syncthreads();   // protect LDS buffers before next prefetch overwrites
  }

  // normalize and emit concat [B*S,128] bf16
  bf16* co = Cc + ((size_t)(b * S + q0 + half * 8) * DM) + h * 32;
#pragma unroll
  for (int v = 0; v < 8; ++v) {
    const float inv = 1.0f / lrow[v];
    co[(size_t)v * DM + l16]      = (bf16)(acc0[v] * inv);
    co[(size_t)v * DM + 16 + l16] = (bf16)(acc1[v] * inv);
  }
}

extern "C" void kernel_launch(void* const* d_in, const int* in_sizes, int n_in,
                              void* d_out, int out_size, void* d_ws, size_t ws_size,
                              hipStream_t stream) {
  const float* q  = (const float*)d_in[0];
  const float* k  = (const float*)d_in[1];
  const float* v  = (const float*)d_in[2];
  const float* Wq = (const float*)d_in[3];
  const float* bq = (const float*)d_in[4];
  const float* Wk = (const float*)d_in[5];
  const float* bk = (const float*)d_in[6];
  const float* Wv = (const float*)d_in[7];
  const float* bv = (const float*)d_in[8];
  const float* Wo = (const float*)d_in[9];
  const float* bo = (const float*)d_in[10];
  float* out = (float*)d_out;

  const int M = in_sizes[0] / DM;   // B*S = 8192
  const int B = 2;
  const int S = M / B;              // 4096

  // workspace: 4 bf16 buffers of M*128 (8 MB total)
  bf16* Qp = (bf16*)d_ws;
  bf16* Kp = Qp + (size_t)M * DM;
  bf16* Vt = Kp + (size_t)M * DM;
  bf16* Cc = Vt + (size_t)M * DM;

  const dim3 blk(256);
  gemm128_kernel<<<dim3(M / 128), blk, 0, stream>>>(q, nullptr, Wq, bq, Qp, nullptr, 0, S);
  gemm128_kernel<<<dim3(M / 128), blk, 0, stream>>>(k, nullptr, Wk, bk, Kp, nullptr, 0, S);
  gemm128_kernel<<<dim3(M / 128), blk, 0, stream>>>(v, nullptr, Wv, bv, Vt, nullptr, 2, S);
  flash_attn_kernel<<<dim3(B * NH, S / 128), blk, 0, stream>>>(Qp, Kp, Vt, Cc, S);
  gemm128_kernel<<<dim3(M / 128), blk, 0, stream>>>(nullptr, Cc, Wo, bo, nullptr, out, 1, S);
}
